// CharRNN_30434138260039
// MI455X (gfx1250) — matrix operand order repaired and checked by reading, not compiled, not run on hardware
//
#include <hip/hip_runtime.h>
#include <hip/hip_bf16.h>
#include <math.h>

// ---------------------------------------------------------------------------
// CharRNN forward on gfx1250 (CDNA5): bf16 WMMA GEMMs with a ping-pong
// software pipeline (no spills), compile-time K, immediate tile offsets,
// L2-resident weights, non-temporal logits stream.
// ---------------------------------------------------------------------------

typedef __attribute__((ext_vector_type(16))) __bf16 bf16x16;
typedef __attribute__((ext_vector_type(8)))  __bf16 bf16x8;
typedef __attribute__((ext_vector_type(8)))  float  f32x8;
typedef __attribute__((ext_vector_type(8)))  float  f32v8;

constexpr int kB = 32, kT = 256, kV = 32000, kD = 512, kH = 2048;
constexpr float kH0 = 0.1f;

// --- WMMA helpers ----------------------------------------------------------
__device__ __forceinline__ f32x8 wmma_bf16(bf16x16 a, bf16x16 b, f32x8 c) {
  return __builtin_amdgcn_wmma_f32_16x16x32_bf16(
      /*neg_a=*/false, a, /*neg_b=*/false, b,
      /*c_mod=*/(short)0, c, /*reuse_a=*/false, /*reuse_b=*/false);
}

__device__ __forceinline__ bf16x16 cat8(bf16x8 lo, bf16x8 hi) {
  bf16x16 r;
#pragma unroll
  for (int i = 0; i < 8; ++i) { r[i] = lo[i]; r[8 + i] = hi[i]; }
  return r;
}

// 16 bf16 elements as two b128 loads at p and p+hiOff (immediate offset).
__device__ __forceinline__ bf16x16 ld16(const __bf16* p, int hiOff) {
  return cat8(*(const bf16x8*)p, *(const bf16x8*)(p + hiOff));
}

// Ping-pong pipelined 16x64 wave tile. 4 f32 accumulators; A reused across
// the 4 WMMAs of each K-step. Two tile stages alternate so loads of one
// stage overlap the WMMAs of the other, with zero register copies.
// Lane pointers (contiguous k = half*16 + s, same order for A and B):
//   pa: A row (m0+lane%16) + half*16;  K-step loads at +0 / +8
//   pb: W row (n0+lane%16) + half*16;  B-tile j at immediate offset j*16*K
template <int K, bool CONST_A>
__device__ __forceinline__ void mm_loop_16x64(const __bf16* pa,
                                              const __bf16* pb,
                                              __bf16 afill, f32x8 acc[4]) {
  static_assert((K % 64) == 0, "need an even number of 32-wide K-steps");
  constexpr int TB = 16 * K;               // elements between B tiles

  bf16x16 ca;
  if (CONST_A) {
#pragma unroll
    for (int i = 0; i < 16; ++i) ca[i] = afill;
  }

  // Stage A: even K-steps, Stage B: odd K-steps.
  bf16x16 aA, b0A, b1A, b2A, b3A;
  bf16x16 aB, b0B, b1B, b2B, b3B;

  if (!CONST_A) aA = ld16(pa, 8);
  b0A = ld16(pb + 0 * TB, 8);
  b1A = ld16(pb + 1 * TB, 8);
  b2A = ld16(pb + 2 * TB, 8);
  b3A = ld16(pb + 3 * TB, 8);

#pragma unroll 1
  for (int k0 = 0; k0 < K; k0 += 64) {
    // Load stage B (k0+32) while stage-A WMMAs run below.
    if (!CONST_A) aB = ld16(pa + 32, 8);
    b0B = ld16(pb + 32 + 0 * TB, 8);
    b1B = ld16(pb + 32 + 1 * TB, 8);
    b2B = ld16(pb + 32 + 2 * TB, 8);
    b3B = ld16(pb + 32 + 3 * TB, 8);
    __builtin_prefetch(pb + 512, 0, 1);    // global_prefetch_b8, ~8 steps out

    acc[0] = wmma_bf16(CONST_A ? ca : aA, b0A, acc[0]);
    acc[1] = wmma_bf16(CONST_A ? ca : aA, b1A, acc[1]);
    acc[2] = wmma_bf16(CONST_A ? ca : aA, b2A, acc[2]);
    acc[3] = wmma_bf16(CONST_A ? ca : aA, b3A, acc[3]);

    // Load stage A for the next iteration (k0+64) under stage-B WMMAs.
    if (k0 + 64 < K) {
      pa += 64; pb += 64;
      if (!CONST_A) aA = ld16(pa, 8);
      b0A = ld16(pb + 0 * TB, 8);
      b1A = ld16(pb + 1 * TB, 8);
      b2A = ld16(pb + 2 * TB, 8);
      b3A = ld16(pb + 3 * TB, 8);
    }

    acc[0] = wmma_bf16(CONST_A ? ca : aB, b0B, acc[0]);
    acc[1] = wmma_bf16(CONST_A ? ca : aB, b1B, acc[1]);
    acc[2] = wmma_bf16(CONST_A ? ca : aB, b2B, acc[2]);
    acc[3] = wmma_bf16(CONST_A ? ca : aB, b3B, acc[3]);
  }
}

// --- Generic GEMM: C[M,N] = A[M,K](bf16) * W[N,K](bf16)^T + bias -----------
// Consecutive waves in a block cover consecutive M-tiles with the SAME B
// columns -> blocks re-read identical W tiles (bf16 V_w = 131 MB fits the
// 192 MB L2). C is streamed with non-temporal stores.
template <int K>
__global__ __launch_bounds__(256, 2)
void gemm_bf16_nt(const __bf16* __restrict__ A,
                  const __bf16* __restrict__ Wt,
                  const float* __restrict__ bias,
                  float* __restrict__ C,
                  int M, int N) {
  const int wave   = blockIdx.x * (blockDim.x >> 5) + (threadIdx.x >> 5);
  const int mtiles = M >> 4;
  const int mt = wave % mtiles;
  const int ng = wave / mtiles;
  const int m0 = mt * 16, n0 = ng * 64;

  const int lane = threadIdx.x & 31;
  const int half = lane >> 4;
  const int nc   = lane & 15;

  const __bf16* pa = A  + (size_t)(m0 + nc) * K + half * 16;
  const __bf16* pb = Wt + (size_t)(n0 + nc) * K + half * 16;

  f32x8 acc[4] = {};
  mm_loop_16x64<K, false>(pa, pb, (__bf16)0.0f, acc);

  // C/D layout: N = lane%16, M = vgpr + 8*(lane/16)
#pragma unroll
  for (int j = 0; j < 4; ++j) {
    const int n  = n0 + 16 * j + nc;
    const float bv = bias ? bias[n] : 0.0f;
#pragma unroll
    for (int v = 0; v < 8; ++v) {
      const int m = m0 + v + 8 * half;
      __builtin_nontemporal_store(acc[j][v] + bv, &C[(size_t)m * N + n]);
    }
  }
}

// --- Elman recurrence step: h_t = tanh(ux_t + h_{t-1} @ W^T) ---------------
// h_all is bf16 [B*T, H], row index b*T + t. One launch per time step.
__global__ __launch_bounds__(256, 2)
void rnn_step(const float* __restrict__ ux,      // [B*T, H] f32
              const __bf16* __restrict__ Wbf,    // [H, H] bf16
              __bf16* __restrict__ h_all,        // [B*T, H] bf16
              int t) {
  const int wave   = blockIdx.x * (blockDim.x >> 5) + (threadIdx.x >> 5);
  const int mtiles = kB >> 4;              // 2
  const int mt = wave % mtiles;
  const int ng = wave / mtiles;
  const int m0 = mt * 16, n0 = ng * 64;

  const int lane = threadIdx.x & 31;
  const int half = lane >> 4;
  const int nc   = lane & 15;
  const size_t rs = (size_t)kT * kH;       // batch-row stride of h_all

  const __bf16* pb = Wbf + (size_t)(n0 + nc) * kH + half * 16;

  f32x8 acc[4] = {};
  if (t == 0) {
    // h_0 = 0.1 * ones: loop-invariant A tile, only W streams.
    mm_loop_16x64<kH, true>(nullptr, pb, (__bf16)kH0, acc);
  } else {
    const __bf16* pa = h_all + (size_t)(t - 1) * kH
                     + (size_t)(m0 + nc) * rs + half * 16;
    mm_loop_16x64<kH, false>(pa, pb, (__bf16)0.0f, acc);
  }

  const float* uxr  = ux    + (size_t)t * kH;
  __bf16*      hout = h_all + (size_t)t * kH;
#pragma unroll
  for (int j = 0; j < 4; ++j) {
    const int n = n0 + 16 * j + nc;
#pragma unroll
    for (int v = 0; v < 8; ++v) {
      const int m = m0 + v + 8 * half;     // batch index
      const float g = tanhf(acc[j][v] + uxr[(size_t)m * rs + n]);
      hout[(size_t)m * rs + n] = (__bf16)g;
    }
  }
}

// --- Conversion / gather kernels (8-wide vectorized) -----------------------
__global__ void f32_to_bf16_v8(const float* __restrict__ src,
                               __bf16* __restrict__ dst, size_t n8) {
  size_t i = (size_t)blockIdx.x * blockDim.x + threadIdx.x;
  const size_t stride = (size_t)gridDim.x * blockDim.x;
  for (; i < n8; i += stride) {
    f32v8 v = *(const f32v8*)(src + i * 8);
    bf16x8 o;
#pragma unroll
    for (int k = 0; k < 8; ++k) o[k] = (__bf16)v[k];
    *(bf16x8*)(dst + i * 8) = o;
  }
}

__global__ void gather_embed_bf16_v8(const float* __restrict__ Ce,  // [V, D]
                                     const int* __restrict__ xb,    // [B*T]
                                     __bf16* __restrict__ emb,      // [B*T, D]
                                     size_t n8) {                   // (B*T*D)/8
  size_t i = (size_t)blockIdx.x * blockDim.x + threadIdx.x;
  const size_t stride = (size_t)gridDim.x * blockDim.x;
  constexpr int D8 = kD / 8;
  for (; i < n8; i += stride) {
    const size_t row = i / D8;
    const int    d8  = (int)(i - row * D8);
    const int    tok = xb[row];
    f32v8 v = *(const f32v8*)(Ce + (size_t)tok * kD + d8 * 8);
    bf16x8 o;
#pragma unroll
    for (int k = 0; k < 8; ++k) o[k] = (__bf16)v[k];
    *(bf16x8*)(emb + i * 8) = o;
  }
}

// --- Loss: -mean over rows of (logit[target] - logsumexp(row)) -------------
__global__ void zero_scalar(float* p) {
  if (blockIdx.x == 0 && threadIdx.x == 0) *p = 0.0f;
}

__global__ void softmax_loss(const float* __restrict__ logits,  // [B*T, V]
                             const int* __restrict__ targets,   // [B*T]
                             float* __restrict__ loss) {
  const int row = blockIdx.x;
  const float* lr = logits + (size_t)row * kV;
  __shared__ float red[256];

  float mx = -INFINITY;
  for (int i = threadIdx.x; i < kV; i += 256)
    mx = fmaxf(mx, __builtin_nontemporal_load(lr + i));
  red[threadIdx.x] = mx;
  __syncthreads();
  for (int s = 128; s > 0; s >>= 1) {
    if ((int)threadIdx.x < s)
      red[threadIdx.x] = fmaxf(red[threadIdx.x], red[threadIdx.x + s]);
    __syncthreads();
  }
  mx = red[0];
  __syncthreads();

  float sum = 0.0f;
  for (int i = threadIdx.x; i < kV; i += 256)
    sum += expf(__builtin_nontemporal_load(lr + i) - mx);
  red[threadIdx.x] = sum;
  __syncthreads();
  for (int s = 128; s > 0; s >>= 1) {
    if ((int)threadIdx.x < s) red[threadIdx.x] += red[threadIdx.x + s];
    __syncthreads();
  }

  if (threadIdx.x == 0) {
    const float lse = mx + logf(red[0]);
    const int tgt = targets[row];
    const float contrib = -(lr[tgt] - lse) / (float)(kB * kT);
    atomicAdd(loss, contrib);
  }
}

// ---------------------------------------------------------------------------
extern "C" void kernel_launch(void* const* d_in, const int* in_sizes, int n_in,
                              void* d_out, int out_size, void* d_ws, size_t ws_size,
                              hipStream_t stream) {
  (void)in_sizes; (void)n_in; (void)out_size; (void)ws_size;

  const int*   xb      = (const int*)  d_in[0];   // [B,T]
  const int*   targets = (const int*)  d_in[1];   // [B,T]
  const float* Ce      = (const float*)d_in[2];   // [V,D]
  const float* U_w     = (const float*)d_in[3];   // [H,D]
  const float* U_b     = (const float*)d_in[4];   // [H]
  const float* W_w     = (const float*)d_in[5];   // [H,H]
  const float* V_w     = (const float*)d_in[6];   // [V,H]
  const float* V_b     = (const float*)d_in[7];   // [V]

  float* logits = (float*)d_out;                       // [B*T, V]
  float* loss   = logits + (size_t)kB * kT * kV;       // scalar

  // Workspace carve-out.
  char* ws = (char*)d_ws;
  size_t off = 0;
  __bf16* emb_bf = (__bf16*)(ws + off); off += (size_t)kB * kT * kD * 2;  //   8 MB
  __bf16* U_bf   = (__bf16*)(ws + off); off += (size_t)kH * kD * 2;       //   2 MB
  __bf16* W_bf   = (__bf16*)(ws + off); off += (size_t)kH * kH * 2;       //   8 MB
  __bf16* V_bf   = (__bf16*)(ws + off); off += (size_t)kV * kH * 2;       // 131 MB
  float*  ux     = (float*) (ws + off); off += (size_t)kB * kT * kH * 4;  //  64 MB
  __bf16* h_all  = (__bf16*)(ws + off); off += (size_t)kB * kT * kH * 2;  //  32 MB

  const int THR = 256;
  auto gsz = [](size_t n, int thr) {
    return (unsigned)((n + (size_t)thr - 1) / (size_t)thr);
  };

  // 1) Precision conversions + embedding gather (8-wide).
  {
    size_t n8;
    n8 = (size_t)kH * kD / 8;
    f32_to_bf16_v8<<<gsz(n8, THR), THR, 0, stream>>>(U_w, U_bf, n8);
    n8 = (size_t)kH * kH / 8;
    f32_to_bf16_v8<<<gsz(n8, THR), THR, 0, stream>>>(W_w, W_bf, n8);
    n8 = (size_t)kV * kH / 8;
    f32_to_bf16_v8<<<gsz(n8, THR), THR, 0, stream>>>(V_w, V_bf, n8);
    n8 = (size_t)kB * kT * kD / 8;
    gather_embed_bf16_v8<<<gsz(n8, THR), THR, 0, stream>>>(Ce, xb, emb_bf, n8);
  }

  // 2) ux = emb @ U_w^T + U_b : [8192,512] x [512,2048]
  {
    const int M = kB * kT, N = kH;
    const int waves = (M / 16) * (N / 64);           // 16384
    gemm_bf16_nt<kD><<<waves / 8, THR, 0, stream>>>(emb_bf, U_bf, U_b, ux, M, N);
  }

  // 3) Recurrence: 256 sequential steps, each [32,2048] x [2048,2048] + tanh.
  {
    const int waves = (kB / 16) * (kH / 64);         // 64 -> 8 blocks
    for (int t = 0; t < kT; ++t)
      rnn_step<<<waves / 8, THR, 0, stream>>>(ux, W_bf, h_all, t);
  }

  // 4) logits = h_all @ V_w^T + V_b : [8192,2048] x [2048,32000]  (~1.07 TFLOP)
  {
    const int M = kB * kT, N = kV;
    const int waves = (M / 16) * (N / 64);           // 256000 -> 32000 blocks
    gemm_bf16_nt<kH><<<waves / 8, THR, 0, stream>>>(h_all, V_bf, V_b, logits, M, N);
  }

  // 5) Loss: per-row logsumexp + gathered target logit, mean over rows.
  zero_scalar<<<1, 32, 0, stream>>>(loss);
  softmax_loss<<<kB * kT, THR, 0, stream>>>(logits, targets, loss);
}
